// GATENet_31679678775980
// MI455X (gfx1250) — matrix operand order repaired and checked by reading, not compiled, not run on hardware
//
#include <hip/hip_runtime.h>
#include <hip/hip_bf16.h>
#include <cmath>

// ---------------------------------------------------------------------------
// Types for CDNA5 WMMA (wave32). 16x16x32 f16 -> f32 accumulate.
// ---------------------------------------------------------------------------
typedef __attribute__((ext_vector_type(16))) _Float16 v16h;
typedef __attribute__((ext_vector_type(8)))  _Float16 v8h;
typedef __attribute__((ext_vector_type(2)))  _Float16 v2h;
typedef __attribute__((ext_vector_type(8)))  float    v8f;

#define HID   128
#define HEADS 2
#define DE    64
#define ENUM_ 64
#define FOUT  64
#define HL    128

// ---------------------------------------------------------------------------
// Device helpers
// ---------------------------------------------------------------------------
__device__ __forceinline__ float sigm(float x) { return 1.f / (1.f + __expf(-x)); }

__device__ __forceinline__ void atomicMaxF(float* addr, float v) {
  // -inf initialized; classic signed/unsigned trick -> global_atomic_{max,min}
  if (v >= 0.f) atomicMax((int*)addr, __float_as_int(v));
  else          atomicMin((unsigned int*)addr, __float_as_uint(v));
}

// A fragment: 16x32 f16, contiguous K (lanes 0-15: rows, K 0..15;
// lanes 16-31: same rows, K 16..31). Loads f32 (b128), converts in regs.
__device__ __forceinline__ v16h load_a_frag(const float* __restrict__ A,
                                            long rowBase, int lda, int k0, int M) {
  int lane = threadIdx.x & 31;
  int m = lane & 15, g = lane >> 4;
  long row = rowBase + m;
  v16h a;
  if (row < (long)M) {
    const float* p = A + row * (long)lda + k0 + g * 16;
#pragma unroll
    for (int j = 0; j < 8; ++j) a[j]     = (_Float16)p[j];
#pragma unroll
    for (int j = 0; j < 8; ++j) a[8 + j] = (_Float16)p[8 + j];
  } else {
#pragma unroll
    for (int j = 0; j < 16; ++j) a[j] = (_Float16)0.f;
  }
  return a;
}

// B fragment from LDS panel stored column-major: Bs[c * KP + k], KP = K + 8
// (halves). Lane (n,g) needs B[k0+g*16 .. +15][cLocal+n]: 16 *contiguous*
// halves -> two ds_load_b128 (16-byte aligned: KP, k0, g*16 all multiples of 8).
__device__ __forceinline__ v16h load_b_frag(const _Float16* __restrict__ Bs,
                                            int KP, int k0, int cLocal) {
  int lane = threadIdx.x & 31;
  int n = lane & 15, g = lane >> 4;
  const _Float16* p = Bs + (size_t)(cLocal + n) * KP + k0 + g * 16;
  v16h b;
  *(v8h*)&b       = *(const v8h*)p;
  *((v8h*)&b + 1) = *(const v8h*)(p + 8);
  return b;
}

// ---------------------------------------------------------------------------
// Generic WMMA GEMM: C[M,Nc] (+)= A[M,K] * B(K,Nc) (+ bias1 + bias2)
// B row-major [K,ldb] (transB=0) or Bt row-major [Nc,ldb=K] (transB=1).
// Block: 128 threads = 4 waves; 64x64 tile; wave computes 32x32 (2x2 WMMA).
// K in {32,64,96,128}. B panel staged f32->f16 column-major in LDS.
// ---------------------------------------------------------------------------
__global__ __launch_bounds__(128) void k_wmma_gemm(
    const float* __restrict__ A, int lda,
    const float* __restrict__ B, int ldb, int transB,
    float* __restrict__ C, int ldc,
    const float* __restrict__ bias1, const float* __restrict__ bias2,
    int M, int Nc, int K, int addC) {
  __shared__ __align__(16) _Float16 Bs[64 * 136];  // max KP = 128 + 8
  const int KP   = K + 8;
  const int tid  = threadIdx.x;
  const int col0 = blockIdx.x * 64;
  const long row0 = (long)blockIdx.y * 64;

  // Stage B panel: packed half-pairs, consecutive threads -> consecutive banks.
  const int Kh = K >> 1;
  for (int i = tid; i < 64 * Kh; i += 128) {
    int c = i / Kh;
    int k = (i - c * Kh) * 2;
    int col = col0 + c;
    float v0 = 0.f, v1 = 0.f;
    if (col < Nc) {
      if (transB) {
        const float* bp = B + (long)col * ldb + k;
        v0 = bp[0]; v1 = bp[1];
      } else {
        v0 = B[(long)k * ldb + col];
        v1 = B[(long)(k + 1) * ldb + col];
      }
    }
    v2h pk; pk[0] = (_Float16)v0; pk[1] = (_Float16)v1;
    *(v2h*)(Bs + (size_t)c * KP + k) = pk;
  }

  const int wid = tid >> 5;
  const int wr = wid >> 1, wc = wid & 1;
  const long rowBase = row0 + wr * 32;

  // One-shot prefetch of the tail of each lane's A row (-> global_prefetch_b8).
  {
    long prow = rowBase + (tid & 31);
    if (prow < (long)M && K > 32) __builtin_prefetch(A + prow * (long)lda + 32, 0, 1);
  }
  __syncthreads();

  v8f acc[2][2] = {};
  for (int k0 = 0; k0 < K; k0 += 32) {
    v16h a0 = load_a_frag(A, rowBase,      lda, k0, M);
    v16h a1 = load_a_frag(A, rowBase + 16, lda, k0, M);
    v16h b0 = load_b_frag(Bs, KP, k0, wc * 32);
    v16h b1 = load_b_frag(Bs, KP, k0, wc * 32 + 16);
    acc[0][0] = __builtin_amdgcn_wmma_f32_16x16x32_f16(false, a0, false, b0, (short)0, acc[0][0], false, false);
    acc[0][1] = __builtin_amdgcn_wmma_f32_16x16x32_f16(false, a0, false, b1, (short)0, acc[0][1], false, false);
    acc[1][0] = __builtin_amdgcn_wmma_f32_16x16x32_f16(false, a1, false, b0, (short)0, acc[1][0], false, false);
    acc[1][1] = __builtin_amdgcn_wmma_f32_16x16x32_f16(false, a1, false, b1, (short)0, acc[1][1], false, false);
  }

  const int lane = tid & 31;
  const int g = lane >> 4, nn = lane & 15;
#pragma unroll
  for (int ti = 0; ti < 2; ++ti)
#pragma unroll
    for (int tj = 0; tj < 2; ++tj) {
      int col = col0 + wc * 32 + tj * 16 + nn;
      if (col >= Nc) continue;
      float badd = (bias1 ? bias1[col] : 0.f) + (bias2 ? bias2[col] : 0.f);
#pragma unroll
      for (int r = 0; r < 8; ++r) {
        long row = rowBase + ti * 16 + g * 8 + r;
        if (row < (long)M) {
          float v = acc[ti][tj][r] + badd;
          float* cp = C + row * (long)ldc + col;
          if (addC) *cp += v; else *cp = v;
        }
      }
    }
}

// ---------------------------------------------------------------------------
// Elementwise / graph kernels
// ---------------------------------------------------------------------------
__global__ void k_fill(float* p, float v, long n) {
  long i = (long)blockIdx.x * blockDim.x + threadIdx.x;
  if (i < n) p[i] = v;
}

__global__ void k_embed(const int* __restrict__ x_idx, const float* __restrict__ x_flt,
                        const float* __restrict__ emb_aa, const float* __restrict__ emb_ss,
                        float* __restrict__ x, int N) {
  long i = (long)blockIdx.x * blockDim.x + threadIdx.x;
  if (i >= (long)N * HID) return;
  int n = (int)(i >> 7), c = (int)(i & 127);
  float v;
  if (c < 123) {
    int aa = x_idx[n * 2 + 0], ss = x_idx[n * 2 + 1];
    v = emb_aa[aa * 123 + c] + emb_ss[ss * 123 + c];
  } else {
    v = x_flt[n * 5 + (c - 123)];
  }
  x[i] = v;
}

// asrc/adst per (node, head): 128-wide dot against att vectors
__global__ void k_att_node(const float* __restrict__ h, const float* __restrict__ att_src,
                           const float* __restrict__ att_dst,
                           float* __restrict__ asrc, float* __restrict__ adst, int N) {
  int i = blockIdx.x * blockDim.x + threadIdx.x;
  if (i >= N * HEADS) return;
  int n = i >> 1, hh = i & 1;
  const float* hp = h + (long)n * (HEADS * HID) + hh * HID;
  float s = 0.f, d = 0.f;
#pragma unroll 4
  for (int c = 0; c < HID; ++c) {
    float hv = hp[c];
    s += hv * att_src[hh * HID + c];
    d += hv * att_dst[hh * HID + c];
  }
  asrc[i] = s; adst[i] = d;
}

// per-attr edge attention term: ae[a][h] = dot(he_tab[a,h,:], att_edge[h,:])
__global__ void k_ae_tab(const float* __restrict__ he_tab, const float* __restrict__ att_edge,
                         float* __restrict__ ae) {
  int i = blockIdx.x * blockDim.x + threadIdx.x;
  if (i >= ENUM_ * HEADS) return;
  int a = i >> 1, hh = i & 1;
  float s = 0.f;
  for (int c = 0; c < HID; ++c) s += he_tab[a * (HEADS * HID) + hh * HID + c] * att_edge[hh * HID + c];
  ae[i] = s;
}

__global__ void k_edge_score(const int* __restrict__ ei, const int* __restrict__ attr,
                             const float* __restrict__ asrc, const float* __restrict__ adst,
                             const float* __restrict__ ae,
                             float* __restrict__ abuf, float* __restrict__ mmax, int E) {
  int e = blockIdx.x * blockDim.x + threadIdx.x;
  if (e >= E) return;
  int s = ei[e], d = ei[E + e], a = attr[e];
#pragma unroll
  for (int hh = 0; hh < HEADS; ++hh) {
    float v = asrc[s * 2 + hh] + adst[d * 2 + hh] + ae[a * 2 + hh];
    v = v > 0.f ? v : 0.2f * v;  // leaky_relu(0.2)
    abuf[(long)e * 2 + hh] = v;
    atomicMaxF(&mmax[d * 2 + hh], v);
  }
}

__global__ void k_edge_exp(const int* __restrict__ ei, float* __restrict__ abuf,
                           const float* __restrict__ mmax, float* __restrict__ zsum, int E) {
  int e = blockIdx.x * blockDim.x + threadIdx.x;
  if (e >= E) return;
  int d = ei[E + e];
#pragma unroll
  for (int hh = 0; hh < HEADS; ++hh) {
    float ex = __expf(abuf[(long)e * 2 + hh] - mmax[d * 2 + hh]);
    abuf[(long)e * 2 + hh] = ex;
    atomicAdd(&zsum[d * 2 + hh], ex);
  }
}

__global__ void k_edge_alpha(const int* __restrict__ ei, float* __restrict__ abuf,
                             const float* __restrict__ zsum, int E) {
  long i = (long)blockIdx.x * blockDim.x + threadIdx.x;
  if (i >= (long)E * 2) return;
  int e = (int)(i >> 1), hh = (int)(i & 1);
  int d = ei[E + e];
  abuf[i] = abuf[i] / (zsum[d * 2 + hh] + 1e-16f);
}

// agg[dst,c] += 0.5 * sum_h alpha_h * (h[src,h,c] + he_tab[attr,h,c])
__global__ void k_edge_aggr(const int* __restrict__ ei, const int* __restrict__ attr,
                            const float* __restrict__ abuf, const float* __restrict__ h,
                            const float* __restrict__ he_tab, float* __restrict__ agg, int E) {
  long i = (long)blockIdx.x * blockDim.x + threadIdx.x;
  if (i >= (long)E * HID) return;
  int e = (int)(i >> 7), c = (int)(i & 127);
  int s = ei[e], d = ei[E + e], a = attr[e];
  float a0 = abuf[(long)e * 2 + 0], a1 = abuf[(long)e * 2 + 1];
  float v = 0.5f * (a0 * (h[(long)s * 256 + c]       + he_tab[a * 256 + c]) +
                    a1 * (h[(long)s * 256 + 128 + c] + he_tab[a * 256 + 128 + c]));
  atomicAdd(&agg[(long)d * HID + c], v);
}

__global__ void k_combine(float* __restrict__ x_l, float* __restrict__ x_r,
                          const float* __restrict__ agg_l, const float* __restrict__ agg_r,
                          const float* __restrict__ bias_l, const float* __restrict__ bias_r,
                          float* __restrict__ xs_l, float* __restrict__ xs_r, int N) {
  long i = (long)blockIdx.x * blockDim.x + threadIdx.x;
  if (i >= (long)N * HID) return;
  int c = (int)(i & 127);
  float xl = x_l[i], xr = x_r[i], mix = 0.5f * (xl + xr);
  float rl = agg_l[i] + bias_l[c]; rl = rl > 0.f ? rl : 0.f;
  float rr = agg_r[i] + bias_r[c]; rr = rr > 0.f ? rr : 0.f;
  float nl = mix + rl, nr = mix + rr;
  x_l[i] = nl; x_r[i] = nr; xs_l[i] = nl; xs_r[i] = nr;
}

// LSTM gate nonlinearity (torch order i,f,g,o in chunks of 128)
__global__ void k_lstm_gate(const float* __restrict__ G, float* __restrict__ cbuf,
                            float* __restrict__ hout, int N, int firstStep) {
  long i = (long)blockIdx.x * blockDim.x + threadIdx.x;
  if (i >= (long)N * HL) return;
  int n = (int)(i >> 7), u = (int)(i & 127);
  const float* gp = G + (long)n * (4 * HL);
  float ig = sigm(gp[u]), fg = sigm(gp[HL + u]);
  float gg = tanhf(gp[2 * HL + u]), og = sigm(gp[3 * HL + u]);
  float cp = firstStep ? 0.f : cbuf[i];
  float c = fg * cp + ig * gg;
  cbuf[i] = c;
  hout[i] = og * tanhf(c);
}

__global__ void k_jk_score(const float* __restrict__ hseq, const float* __restrict__ attW,
                           const float* __restrict__ attB, float* __restrict__ score, int N) {
  int i = blockIdx.x * blockDim.x + threadIdx.x;
  if (i >= N * 3) return;
  int n = i / 3, l = i % 3;
  const float* hf = hseq + ((long)l * N + n) * HL;
  const float* hb = hseq + ((long)(3 + l) * N + n) * HL;
  float s = attB[0];
#pragma unroll 4
  for (int c = 0; c < HL; ++c) s += hf[c] * attW[c] + hb[c] * attW[HL + c];
  score[(long)n * 3 + l] = s;
}

__global__ void k_jk_combine(const float* __restrict__ xs, const float* __restrict__ score,
                             float* __restrict__ jk, int N) {
  long i = (long)blockIdx.x * blockDim.x + threadIdx.x;
  if (i >= (long)N * HID) return;
  int n = (int)(i >> 7);
  float s0 = score[(long)n * 3], s1 = score[(long)n * 3 + 1], s2 = score[(long)n * 3 + 2];
  float m = fmaxf(s0, fmaxf(s1, s2));
  float e0 = __expf(s0 - m), e1 = __expf(s1 - m), e2 = __expf(s2 - m);
  float inv = 1.f / (e0 + e1 + e2);
  long NL = (long)N * HID;
  jk[i] = (e0 * xs[i] + e1 * xs[NL + i] + e2 * xs[2 * NL + i]) * inv;
}

__global__ void k_gate4(const float* __restrict__ feats, const float* __restrict__ gW,
                        float* __restrict__ gates, int N) {
  int n = blockIdx.x * blockDim.x + threadIdx.x;
  if (n >= N) return;
  long stride = (long)N * FOUT;
  float g[4];
#pragma unroll
  for (int i = 0; i < 4; ++i) {
    const float* fp = feats + i * stride + (long)n * FOUT;
    float s = 0.f;
    for (int c = 0; c < FOUT; ++c) s += fp[c] * gW[c];
    g[i] = s;
  }
  float m = fmaxf(fmaxf(g[0], g[1]), fmaxf(g[2], g[3]));
  float e[4], z = 0.f;
#pragma unroll
  for (int i = 0; i < 4; ++i) { e[i] = __expf(g[i] - m); z += e[i]; }
#pragma unroll
  for (int i = 0; i < 4; ++i) gates[(long)n * 4 + i] = e[i] / z;
}

__global__ void k_final_out(const float* __restrict__ feats, const float* __restrict__ gates,
                            float* __restrict__ out, int N) {
  long i = (long)blockIdx.x * blockDim.x + threadIdx.x;
  if (i >= (long)N * FOUT) return;
  int n = (int)(i >> 6);
  long stride = (long)N * FOUT;
  float v = 0.f;
#pragma unroll
  for (int k = 0; k < 4; ++k) v += gates[(long)n * 4 + k] * feats[k * stride + i];
  out[i] = v;
}

// ---------------------------------------------------------------------------
// Host orchestration
// ---------------------------------------------------------------------------
static inline void gemm(hipStream_t s, const float* A, int lda, const float* B, int ldb,
                        int transB, float* C, int ldc, const float* b1, const float* b2,
                        int M, int Nc, int K, int addC) {
  dim3 g((unsigned)((Nc + 63) / 64), (unsigned)((M + 63) / 64));
  k_wmma_gemm<<<g, 128, 0, s>>>(A, lda, B, ldb, transB, C, ldc, b1, b2, M, Nc, K, addC);
}
static inline dim3 g1(long n, int b = 256) { return dim3((unsigned)((n + b - 1) / b)); }

extern "C" void kernel_launch(void* const* d_in, const int* in_sizes, int n_in,
                              void* d_out, int out_size, void* d_ws, size_t ws_size,
                              hipStream_t stream) {
  // ---- input mapping: jax pytree flatten (dicts sorted by key) ----
  auto F = [&](int i) { return (const float*)d_in[i]; };
  auto I = [&](int i) { return (const int*)d_in[i]; };
  const int BL = 9, BR = 48;                // branch leaf bases ('l','r')
#define CONV_W(b,i)    F((b) + 0 + 6*(i))   // [128,256]
#define CONV_WE(b,i)   F((b) + 1 + 6*(i))   // [64,256]
#define CONV_ADST(b,i) F((b) + 2 + 6*(i))   // [2,128]
#define CONV_AEDG(b,i) F((b) + 3 + 6*(i))
#define CONV_ASRC(b,i) F((b) + 4 + 6*(i))
#define CONV_B(b,i)    F((b) + 5 + 6*(i))   // [128]
#define DUMMY_W(b)     F((b) + 18)
#define DUMMY_B(b)     F((b) + 19)
#define ENET_W(b,i)    F((b) + 20 + 2*(i))
#define ENET_B(b,i)    F((b) + 21 + 2*(i))
#define EMB_AA(b)      F((b) + 26)
#define EMB_E(b)       F((b) + 27)
#define EMB_SS(b)      F((b) + 28)
#define WHH_B(b) F((b)+29)
#define WHH_F(b) F((b)+30)
#define WIH_B(b) F((b)+31)
#define WIH_F(b) F((b)+32)
#define ATT_W(b) F((b)+33)
#define ATT_B(b) F((b)+34)
#define BHH_B(b) F((b)+35)
#define BHH_F(b) F((b)+36)
#define BIH_B(b) F((b)+37)
#define BIH_F(b) F((b)+38)
  const float* gateW  = F(0);
  const int*   x_idx  = I(87);
  const float* x_flt  = F(88);
  const int*   ei_l   = I(89);
  const int*   ei_r   = I(90);
  const int*   attr_l = I(91);
  const int*   attr_r = I(92);
  const int N = in_sizes[88] / 5;
  const int E = in_sizes[91];
  const long NL = (long)N * HID;

  // ---- workspace layout ----
  float* W = (float*)d_ws;
  size_t off = 0;
  auto alloc = [&](size_t n) { float* p = W + off; off += n; return p; };
  float* x_l   = alloc(NL);      float* x_r   = alloc(NL);
  float* xs_l  = alloc(3 * NL);  float* xs_r  = alloc(3 * NL);
  float* hbuf  = alloc((long)N * 256);
  float* agg_l = alloc(NL);      float* agg_r = alloc(NL);
  float* asrc  = alloc((long)N * 2); float* adst = alloc((long)N * 2);
  float* mmax  = alloc((long)N * 2); float* zsum = alloc((long)N * 2);
  float* abuf  = alloc((long)E * 2);
  float* heTab = alloc(ENUM_ * 256); float* aeTab = alloc(ENUM_ * 2);
  float* etpA_l = alloc(64 * 64);  float* etpB_l = alloc(64 * 64);
  float* etpA_r = alloc(64 * 64);  float* etpB_r = alloc(64 * 64);
  float* G     = alloc((long)N * 512);
  float* cbuf  = alloc(NL);
  float* hseq  = alloc(6 * NL);
  float* score = alloc((long)N * 3);
  float* jk_l  = alloc(NL); float* jk_r = alloc(NL);
  float* sq_l  = alloc(NL); float* sq_r = alloc(NL);
  float* feats = alloc((long)N * FOUT * 4);
  float* gates = alloc((long)N * 4);
  (void)ws_size; (void)n_in; (void)out_size;

  // ---- node features ----
  k_embed<<<g1(NL), 256, 0, stream>>>(x_idx, x_flt, EMB_AA(BL), EMB_SS(BL), x_l, N);
  k_embed<<<g1(NL), 256, 0, stream>>>(x_idx, x_flt, EMB_AA(BR), EMB_SS(BR), x_r, N);

  // ---- GNN layers ----
  const float* etab_l = EMB_E(BL);
  const float* etab_r = EMB_E(BR);
  for (int layer = 0; layer < 3; ++layer) {
    struct Side { int b; const int* ei; const int* attr; const float** etab;
                  float* etpA; float* etpB; float* x; float* agg; };
    Side sides[2] = {{BL, ei_l, attr_l, &etab_l, etpA_l, etpB_l, x_l, agg_l},
                     {BR, ei_r, attr_r, &etab_r, etpA_r, etpB_r, x_r, agg_r}};
    for (int s = 0; s < 2; ++s) {
      Side& sd = sides[s];
      // h = x @ W  -> [N, 2*128]  (WMMA)
      gemm(stream, sd.x, HID, CONV_W(sd.b, layer), 256, 0, hbuf, 256,
           nullptr, nullptr, N, 256, HID, 0);
      // he_tab = etab @ We (64x256, WMMA) — algebraic hoist of per-edge gather-GEMM
      gemm(stream, *sd.etab, DE, CONV_WE(sd.b, layer), 256, 0, heTab, 256,
           nullptr, nullptr, ENUM_, 256, DE, 0);
      k_ae_tab<<<1, 128, 0, stream>>>(heTab, CONV_AEDG(sd.b, layer), aeTab);
      k_att_node<<<g1((long)N * 2), 256, 0, stream>>>(hbuf, CONV_ASRC(sd.b, layer),
                                                      CONV_ADST(sd.b, layer), asrc, adst, N);
      k_fill<<<g1((long)N * 2), 256, 0, stream>>>(mmax, -INFINITY, (long)N * 2);
      k_fill<<<g1((long)N * 2), 256, 0, stream>>>(zsum, 0.f, (long)N * 2);
      k_fill<<<g1(NL), 256, 0, stream>>>(sd.agg, 0.f, NL);
      k_edge_score<<<g1(E), 256, 0, stream>>>(sd.ei, sd.attr, asrc, adst, aeTab, abuf, mmax, E);
      k_edge_exp  <<<g1(E), 256, 0, stream>>>(sd.ei, abuf, mmax, zsum, E);
      k_edge_alpha<<<g1((long)E * 2), 256, 0, stream>>>(sd.ei, abuf, zsum, E);
      k_edge_aggr <<<g1((long)E * HID), 256, 0, stream>>>(sd.ei, sd.attr, abuf, hbuf, heTab,
                                                          sd.agg, E);
      // edgenet update (used by next layer)
      if (layer < 2) {
        float* dst = (layer & 1) ? sd.etpB : sd.etpA;
        gemm(stream, *sd.etab, DE, ENET_W(sd.b, layer), DE, 0, dst, DE,
             ENET_B(sd.b, layer), nullptr, ENUM_, DE, DE, 0);
        *sd.etab = dst;
      }
    }
    k_combine<<<g1(NL), 256, 0, stream>>>(x_l, x_r, agg_l, agg_r,
                                          CONV_B(BL, layer), CONV_B(BR, layer),
                                          xs_l + (long)layer * NL, xs_r + (long)layer * NL, N);
  }

  // ---- JK bidirectional LSTM + layer attention, then dummy projection ----
  struct JB { int b; float* xs; float* jk; float* sq; };
  JB jbs[2] = {{BL, xs_l, jk_l, sq_l}, {BR, xs_r, jk_r, sq_r}};
  for (int s = 0; s < 2; ++s) {
    JB& j = jbs[s];
    // forward
    float* hprev = nullptr;
    for (int t = 0; t < 3; ++t) {
      gemm(stream, j.xs + (long)t * NL, HID, WIH_F(j.b), HID, 1, G, 512,
           BIH_F(j.b), BHH_F(j.b), N, 512, HID, 0);
      if (hprev) gemm(stream, hprev, HL, WHH_F(j.b), HL, 1, G, 512,
                      nullptr, nullptr, N, 512, HL, 1);
      k_lstm_gate<<<g1(NL), 256, 0, stream>>>(G, cbuf, hseq + (long)t * NL, N, hprev == nullptr);
      hprev = hseq + (long)t * NL;
    }
    // backward
    hprev = nullptr;
    for (int tt = 0; tt < 3; ++tt) {
      int t = 2 - tt;
      gemm(stream, j.xs + (long)t * NL, HID, WIH_B(j.b), HID, 1, G, 512,
           BIH_B(j.b), BHH_B(j.b), N, 512, HID, 0);
      if (hprev) gemm(stream, hprev, HL, WHH_B(j.b), HL, 1, G, 512,
                      nullptr, nullptr, N, 512, HL, 1);
      k_lstm_gate<<<g1(NL), 256, 0, stream>>>(G, cbuf, hseq + (long)(3 + t) * NL, N,
                                              hprev == nullptr);
      hprev = hseq + (long)(3 + t) * NL;
    }
    k_jk_score  <<<g1((long)N * 3), 256, 0, stream>>>(hseq, ATT_W(j.b), ATT_B(j.b), score, N);
    k_jk_combine<<<g1(NL), 256, 0, stream>>>(j.xs, score, j.jk, N);
    gemm(stream, j.jk, HID, DUMMY_W(j.b), HID, 0, j.sq, HID,
         DUMMY_B(j.b), nullptr, N, HID, HID, 0);
  }

  // ---- final heads + softmax gate ----
  const float* zins[4] = {jk_l, sq_l, jk_r, sq_r};
  for (int i = 0; i < 4; ++i) {
    gemm(stream, zins[i], HID, F(1 + 2 * i), FOUT, 0, feats + (long)i * N * FOUT, FOUT,
         F(2 + 2 * i), nullptr, N, FOUT, HID, 0);
  }
  k_gate4<<<g1(N), 256, 0, stream>>>(feats, gateW, gates, N);
  k_final_out<<<g1((long)N * FOUT), 256, 0, stream>>>(feats, gates, (float*)d_out, N);
}
